// VCLLayer_37752762532123
// MI455X (gfx1250) — compile-verified, matrix operand order checked by reading, not run on hardware
//
#include <hip/hip_runtime.h>

typedef __attribute__((ext_vector_type(16))) _Float16 v16h;
typedef __attribute__((ext_vector_type(8)))  _Float16 v8h;
typedef __attribute__((ext_vector_type(8)))  float    v8f;

#define K_SAMP  64
#define M_DIM   256
#define IN_DIM  1024
#define OUT_DIM 1024

#define O_TILE  64
#define NK      32          // n-depth staged in LDS per iteration (= one WMMA K)
#define XSTR    40          // padded row stride (halves): 80B rows -> 16B-aligned, conflict-free b128
#define WSTR    40

__device__ __forceinline__ unsigned int pack2h(_Float16 a, _Float16 b) {
    union { _Float16 h[2]; unsigned int u; } p;
    p.h[0] = a; p.h[1] = b;
    return p.u;
}

// A/B fragment for v_wmma_f32_16x16x32_f16, per ISA 16-bit layout:
// per lane: K = half*8..half*8+7 (VGPR0-3) and K = 16+half*8..+7 (VGPR4-7)
__device__ __forceinline__ v16h load_frag(const _Float16* p) {
    v8h lo = *(const v8h*)(p);        // 16B, aligned
    v8h hi = *(const v8h*)(p + 16);   // +32B
    return __builtin_shufflevector(lo, hi, 0,1,2,3,4,5,6,7,8,9,10,11,12,13,14,15);
}

__global__ __launch_bounds__(256)
void vcl_bayes_gemm_f16x2(const float* __restrict__ x,     // [K, M, IN]
                          const float* __restrict__ wmu,   // [IN, OUT]
                          const float* __restrict__ wlv,   // [IN, OUT]
                          const float* __restrict__ bmu,   // [OUT]
                          const float* __restrict__ blv,   // [OUT]
                          const float* __restrict__ weps,  // [K, IN, OUT]
                          const float* __restrict__ beps,  // [K, OUT]
                          float* __restrict__ out)         // [K, M, OUT]
{
    __shared__ _Float16 lds_xh[M_DIM * XSTR];   // 20 KB
    __shared__ _Float16 lds_xl[M_DIM * XSTR];   // 20 KB
    __shared__ _Float16 lds_wh[O_TILE * WSTR];  // 5 KB   (transposed: [o][n])
    __shared__ _Float16 lds_wl[O_TILE * WSTR];  // 5 KB

    const int tid   = threadIdx.x;
    const int lane  = tid & 31;
    const int wave  = tid >> 5;          // 0..7: wave owns m rows [wave*32, wave*32+32)
    const int half  = lane >> 4;         // selects K-half inside a WMMA fragment
    const int l16   = lane & 15;

    const int k     = blockIdx.y;
    const int obase = blockIdx.x * O_TILE;

    const float* __restrict__ xk   = x    + (size_t)k * M_DIM * IN_DIM;
    const float* __restrict__ epsk = weps + (size_t)k * IN_DIM * OUT_DIM;

    v8f acc[2][4];
    #pragma unroll
    for (int mt = 0; mt < 2; ++mt)
        #pragma unroll
        for (int ot = 0; ot < 4; ++ot)
            #pragma unroll
            for (int e = 0; e < 8; ++e)
                acc[mt][ot][e] = 0.0f;

    const int mrow = wave * 32;

    for (int nc = 0; nc < IN_DIM / NK; ++nc) {
        const int nb = nc * NK;
        __syncthreads();   // previous iteration done reading LDS

        // ---- stage x tile, split hi/lo fp16: thread t fills row m = t (32 values) ----
        {
            const float* gx = xk + (size_t)tid * IN_DIM + nb;
            #pragma unroll
            for (int j = 0; j < 8; ++j) {
                float4 v = *(const float4*)(gx + j * 4);
                _Float16 h0 = (_Float16)v.x, h1 = (_Float16)v.y;
                _Float16 h2 = (_Float16)v.z, h3 = (_Float16)v.w;
                _Float16 l0 = (_Float16)(v.x - (float)h0);
                _Float16 l1 = (_Float16)(v.y - (float)h1);
                _Float16 l2 = (_Float16)(v.z - (float)h2);
                _Float16 l3 = (_Float16)(v.w - (float)h3);
                *(unsigned int*)&lds_xh[tid * XSTR + j * 4    ] = pack2h(h0, h1);
                *(unsigned int*)&lds_xh[tid * XSTR + j * 4 + 2] = pack2h(h2, h3);
                *(unsigned int*)&lds_xl[tid * XSTR + j * 4    ] = pack2h(l0, l1);
                *(unsigned int*)&lds_xl[tid * XSTR + j * 4 + 2] = pack2h(l2, l3);
            }
        }

        // ---- stage reconstructed weight tile W = mu + exp(lv/2)*eps, split hi/lo, transposed ----
        #pragma unroll
        for (int j = 0; j < 4; ++j) {
            const int p  = tid + 256 * j;      // pair id 0..1023
            const int n2 = p >> 6;             // 0..15 -> n = 2*n2, 2*n2+1
            const int o  = p & 63;
            const int n  = n2 * 2;
            const size_t g0 = (size_t)(nb + n) * OUT_DIM + obase + o;
            const size_t g1 = g0 + OUT_DIM;
            const float w0 = fmaf(__expf(0.5f * wlv[g0]), epsk[g0], wmu[g0]);
            const float w1 = fmaf(__expf(0.5f * wlv[g1]), epsk[g1], wmu[g1]);
            _Float16 h0 = (_Float16)w0, h1 = (_Float16)w1;
            _Float16 l0 = (_Float16)(w0 - (float)h0);
            _Float16 l1 = (_Float16)(w1 - (float)h1);
            *(unsigned int*)&lds_wh[o * WSTR + n] = pack2h(h0, h1);
            *(unsigned int*)&lds_wl[o * WSTR + n] = pack2h(l0, l1);
        }

        // ---- prefetch next chunk's eps stream (the HBM-bound operand) ----
        if (nc + 1 < IN_DIM / NK) {
            #pragma unroll
            for (int j = 0; j < 4; ++j) {
                const int p  = tid + 256 * j;
                const int n  = (p >> 6) * 2;
                const int o  = p & 63;
                const float* q = epsk + (size_t)(nb + NK + n) * OUT_DIM + obase + o;
                __builtin_prefetch(q, 0, 0);
                __builtin_prefetch(q + OUT_DIM, 0, 0);
            }
        }

        __syncthreads();   // tiles visible

        // ---- fragments: 2 m-tiles x {hi,lo}, 4 o-tiles x {hi,lo} ----
        v16h ah[2], al[2], bh[4], bl[4];
        #pragma unroll
        for (int mt = 0; mt < 2; ++mt) {
            ah[mt] = load_frag(&lds_xh[(mrow + mt * 16 + l16) * XSTR + half * 8]);
            al[mt] = load_frag(&lds_xl[(mrow + mt * 16 + l16) * XSTR + half * 8]);
        }
        #pragma unroll
        for (int ot = 0; ot < 4; ++ot) {
            bh[ot] = load_frag(&lds_wh[(ot * 16 + l16) * WSTR + half * 8]);
            bl[ot] = load_frag(&lds_wl[(ot * 16 + l16) * WSTR + half * 8]);
        }

        // ---- 3-pass split accumulation: hh, hl, lh (lo*lo dropped, ~2^-22) ----
        #pragma unroll
        for (int mt = 0; mt < 2; ++mt)
            #pragma unroll
            for (int ot = 0; ot < 4; ++ot)
                acc[mt][ot] = __builtin_amdgcn_wmma_f32_16x16x32_f16(
                    false, ah[mt], false, bh[ot], (short)0, acc[mt][ot], false, false);
        #pragma unroll
        for (int mt = 0; mt < 2; ++mt)
            #pragma unroll
            for (int ot = 0; ot < 4; ++ot)
                acc[mt][ot] = __builtin_amdgcn_wmma_f32_16x16x32_f16(
                    false, ah[mt], false, bl[ot], (short)0, acc[mt][ot], false, false);
        #pragma unroll
        for (int mt = 0; mt < 2; ++mt)
            #pragma unroll
            for (int ot = 0; ot < 4; ++ot)
                acc[mt][ot] = __builtin_amdgcn_wmma_f32_16x16x32_f16(
                    false, al[mt], false, bh[ot], (short)0, acc[mt][ot], false, false);
    }

    // ---- epilogue: reparameterized bias + store ----
    float bterm[4];
    #pragma unroll
    for (int ot = 0; ot < 4; ++ot) {
        const int o = obase + ot * 16 + l16;
        bterm[ot] = fmaf(__expf(0.5f * blv[o]), beps[(size_t)k * OUT_DIM + o], bmu[o]);
    }

    #pragma unroll
    for (int mt = 0; mt < 2; ++mt) {
        #pragma unroll
        for (int ot = 0; ot < 4; ++ot) {
            const int o = obase + ot * 16 + l16;
            #pragma unroll
            for (int v = 0; v < 8; ++v) {
                const int m = mrow + mt * 16 + half * 8 + v;  // C layout: VGPR v -> M=v / M=8+v
                out[((size_t)k * M_DIM + m) * OUT_DIM + o] = acc[mt][ot][v] + bterm[ot];
            }
        }
    }
}

extern "C" void kernel_launch(void* const* d_in, const int* in_sizes, int n_in,
                              void* d_out, int out_size, void* d_ws, size_t ws_size,
                              hipStream_t stream) {
    const float* x    = (const float*)d_in[0];
    const float* wmu  = (const float*)d_in[1];
    const float* wlv  = (const float*)d_in[2];
    const float* bmu  = (const float*)d_in[3];
    const float* blv  = (const float*)d_in[4];
    const float* weps = (const float*)d_in[5];
    const float* beps = (const float*)d_in[6];
    float* out        = (float*)d_out;

    dim3 grid(OUT_DIM / O_TILE, K_SAMP);   // (16, 64); o-tiles fast so one k's tiles share x[k]/mu in L2
    vcl_bayes_gemm_f16x2<<<grid, 256, 0, stream>>>(x, wmu, wlv, bmu, blv, weps, beps, out);
}
